// GATv2_26877905339096
// MI455X (gfx1250) — compile-verified, matrix operand order checked by reading, not run on hardware
//
#include <hip/hip_runtime.h>

// ---------------------------------------------------------------------------
// GATv2 link predictor for MI455X (gfx1250, wave32, WMMA)
// ---------------------------------------------------------------------------

typedef __attribute__((ext_vector_type(16))) _Float16 v16h;
typedef __attribute__((ext_vector_type(8)))  _Float16 v8h;
typedef __attribute__((ext_vector_type(8)))  float    v8f;

#define N_NODES 50000
#define E_EDGES 800000
#define P_CAND  100000
#define HDIM    128          // H*D = 2*64
#define NEG_SLOPE 0.2f

// ----- helpers --------------------------------------------------------------

__device__ __forceinline__ int f2ord(float f) {
    int i = __float_as_int(f);
    return i >= 0 ? i : (i ^ 0x7fffffff);
}
__device__ __forceinline__ float ord2f(int i) {
    return __int_as_float(i >= 0 ? i : (i ^ 0x7fffffff));
}

// Build A-fragment (16x32 f16, per ISA layout: elems 0..7 = K k0..k0+7,
// elems 8..15 = K k0+16..k0+23) from f32 row data.
__device__ __forceinline__ v16h load_a_f32(const float* __restrict__ p) {
    float4 a0 = *(const float4*)(p);
    float4 a1 = *(const float4*)(p + 4);
    float4 a2 = *(const float4*)(p + 16);
    float4 a3 = *(const float4*)(p + 20);
    v16h v;
    v[0]=(_Float16)a0.x; v[1]=(_Float16)a0.y; v[2]=(_Float16)a0.z; v[3]=(_Float16)a0.w;
    v[4]=(_Float16)a1.x; v[5]=(_Float16)a1.y; v[6]=(_Float16)a1.z; v[7]=(_Float16)a1.w;
    v[8]=(_Float16)a2.x; v[9]=(_Float16)a2.y; v[10]=(_Float16)a2.z; v[11]=(_Float16)a2.w;
    v[12]=(_Float16)a3.x; v[13]=(_Float16)a3.y; v[14]=(_Float16)a3.z; v[15]=(_Float16)a3.w;
    return v;
}

// A-fragment from elementwise product of two f32 rows (predictor z = hs*hd)
__device__ __forceinline__ v16h load_a_prod(const float* __restrict__ sp,
                                            const float* __restrict__ dp) {
    float4 s0 = *(const float4*)(sp),      d0 = *(const float4*)(dp);
    float4 s1 = *(const float4*)(sp + 4),  d1 = *(const float4*)(dp + 4);
    float4 s2 = *(const float4*)(sp + 16), d2 = *(const float4*)(dp + 16);
    float4 s3 = *(const float4*)(sp + 20), d3 = *(const float4*)(dp + 20);
    v16h v;
    v[0]=(_Float16)(s0.x*d0.x); v[1]=(_Float16)(s0.y*d0.y);
    v[2]=(_Float16)(s0.z*d0.z); v[3]=(_Float16)(s0.w*d0.w);
    v[4]=(_Float16)(s1.x*d1.x); v[5]=(_Float16)(s1.y*d1.y);
    v[6]=(_Float16)(s1.z*d1.z); v[7]=(_Float16)(s1.w*d1.w);
    v[8]=(_Float16)(s2.x*d2.x); v[9]=(_Float16)(s2.y*d2.y);
    v[10]=(_Float16)(s2.z*d2.z); v[11]=(_Float16)(s2.w*d2.w);
    v[12]=(_Float16)(s3.x*d3.x); v[13]=(_Float16)(s3.y*d3.y);
    v[14]=(_Float16)(s3.z*d3.z); v[15]=(_Float16)(s3.w*d3.w);
    return v;
}

// B-fragment (32x16 f16, column per lane) from pre-transposed f16 weights
__device__ __forceinline__ v16h load_b_f16(const _Float16* __restrict__ p) {
    v8h lo = *(const v8h*)(p);
    v8h hi = *(const v8h*)(p + 16);
    v16h v;
#pragma unroll
    for (int i = 0; i < 8; ++i) { v[i] = lo[i]; v[8 + i] = hi[i]; }
    return v;
}

// ----- weight prep: transpose + f32->f16 ------------------------------------
__global__ void prep_weights(const float* __restrict__ W0, const float* __restrict__ W1,
                             const float* __restrict__ Wp1, const float* __restrict__ Wp2,
                             _Float16* __restrict__ W0t, _Float16* __restrict__ W1t,
                             _Float16* __restrict__ Wp1t, _Float16* __restrict__ Wp2t) {
    int idx = blockIdx.x * blockDim.x + threadIdx.x;
    if (idx < 128 * 128) {              // W0,W1: [K=128][N=128] -> t[N][K]
        int c = idx >> 7, k = idx & 127;
        W0t[idx] = (_Float16)W0[k * 128 + c];
        W1t[idx] = (_Float16)W1[k * 128 + c];
    }
    if (idx < 64 * 128) {               // Wp1: [K=128][N=64] -> t[64][128]
        int c = idx >> 7, k = idx & 127;
        Wp1t[idx] = (_Float16)Wp1[k * 64 + c];
    }
    if (idx < 64 * 64) {                // Wp2: [K=64][N=64] -> t[64][64]
        int c = idx >> 6, k = idx & 63;
        Wp2t[idx] = (_Float16)Wp2[k * 64 + c];
    }
}

// ----- C[M,128] = A[M,128] @ W (Bt = f16 transposed weights) ----------------
__global__ __launch_bounds__(256)
void gemm128_wmma(const float* __restrict__ A, const _Float16* __restrict__ Bt,
                  float* __restrict__ C, int M) {
    int wid  = blockIdx.x * (blockDim.x >> 5) + (threadIdx.x >> 5);
    int lane = threadIdx.x & 31;
    int half = lane >> 4, r = lane & 15;
    int tm = wid >> 3, tn = wid & 7;          // 8 tiles along N=128
    if (tm * 16 >= M) return;
    int row = tm * 16 + r;
    int col = tn * 16 + r;
    v8f acc = {};
#pragma unroll
    for (int kk = 0; kk < 128; kk += 32) {
        int k0 = kk + half * 8;
        v16h a = load_a_f32(A + row * 128 + k0);
        v16h b = load_b_f16(Bt + col * 128 + k0);
        acc = __builtin_amdgcn_wmma_f32_16x16x32_f16(false, a, false, b,
                                                     (short)0, acc, false, false);
    }
    // C layout: VGPR v, lane -> M = v + half*8, N = r
    float* cp = C + (tm * 16 + half * 8) * 128 + tn * 16 + r;
#pragma unroll
    for (int v = 0; v < 8; ++v) cp[v * 128] = acc[v];
}

// ----- edge pass 1: score + segment max (wave per edge) ---------------------
__global__ __launch_bounds__(256)
void edge_score_kernel(const float* __restrict__ feat, const int* __restrict__ src,
                       const int* __restrict__ dst, const float* __restrict__ attn,
                       float* __restrict__ score, int* __restrict__ mOrd) {
    int gid = blockIdx.x * blockDim.x + threadIdx.x;
    int e = gid >> 5;
    if (e >= E_EDGES) return;
    int lane = threadIdx.x & 31;
    int hh = lane >> 4, d4 = (lane & 15) << 2;   // lane -> (head, d-quad)
    int s_ = src[e], d_ = dst[e];
    float4 el = *(const float4*)(feat + s_ * HDIM + hh * 64 + d4);
    float4 er = *(const float4*)(feat + d_ * HDIM + hh * 64 + d4);
    float4 at = *(const float4*)(attn + hh * 64 + d4);
    float x0 = el.x + er.x; x0 = x0 > 0.f ? x0 : NEG_SLOPE * x0;
    float x1 = el.y + er.y; x1 = x1 > 0.f ? x1 : NEG_SLOPE * x1;
    float x2 = el.z + er.z; x2 = x2 > 0.f ? x2 : NEG_SLOPE * x2;
    float x3 = el.w + er.w; x3 = x3 > 0.f ? x3 : NEG_SLOPE * x3;
    float part = x0 * at.x + x1 * at.y + x2 * at.z + x3 * at.w;
    part += __shfl_xor(part, 1, 32);
    part += __shfl_xor(part, 2, 32);
    part += __shfl_xor(part, 4, 32);
    part += __shfl_xor(part, 8, 32);             // reduce within 16-lane half
    if ((lane & 15) == 0) {
        score[e * 2 + hh] = part;
        atomicMax(&mOrd[d_ * 2 + hh], f2ord(part));
    }
}

// ----- edge pass 2: p = exp(score - m[dst]); segment sum --------------------
__global__ __launch_bounds__(256)
void edge_expsum_kernel(const int* __restrict__ dst, const int* __restrict__ mOrd,
                        float* __restrict__ score, float* __restrict__ ssum) {
    int idx = blockIdx.x * blockDim.x + threadIdx.x;
    if (idx >= E_EDGES * 2) return;
    int e = idx >> 1, hh = idx & 1;
    int d_ = dst[e];
    float m = ord2f(mOrd[d_ * 2 + hh]);
    float p = __expf(score[idx] - m);
    score[idx] = p;                  // reuse buffer for p
    atomicAdd(&ssum[d_ * 2 + hh], p);
}

// ----- edge pass 3: rst[dst] += feat[src] * (p / s[dst]) --------------------
__global__ __launch_bounds__(256)
void edge_aggregate_kernel(const float* __restrict__ feat, const int* __restrict__ src,
                           const int* __restrict__ dst, const float* __restrict__ p,
                           const float* __restrict__ ssum, float* __restrict__ rst) {
    int gid = blockIdx.x * blockDim.x + threadIdx.x;
    int e = gid >> 5;
    if (e >= E_EDGES) return;
    int lane = threadIdx.x & 31;
    int hh = lane >> 4, d4 = (lane & 15) << 2;
    int s_ = src[e], d_ = dst[e];
    float a = p[e * 2 + hh] / ssum[d_ * 2 + hh];
    float4 el = *(const float4*)(feat + s_ * HDIM + hh * 64 + d4);
    float* rp = rst + d_ * HDIM + hh * 64 + d4;
    atomicAdd(rp + 0, el.x * a);
    atomicAdd(rp + 1, el.y * a);
    atomicAdd(rp + 2, el.z * a);
    atomicAdd(rp + 3, el.w * a);
}

// ----- residual + ELU -------------------------------------------------------
__global__ __launch_bounds__(256)
void finalize_kernel(const float* __restrict__ rst, const float* __restrict__ hin,
                     float* __restrict__ hout, int residual, int n) {
    int idx = blockIdx.x * blockDim.x + threadIdx.x;
    if (idx >= n) return;
    float v = rst[idx] + (residual ? hin[idx] : 0.0f);
    hout[idx] = v > 0.f ? v : (__expf(v) - 1.0f);
}

// ----- predictor: wave per 16 candidates, 3-layer MLP via WMMA --------------
__global__ __launch_bounds__(128)
void predictor_wmma(const float* __restrict__ h,
                    const int* __restrict__ ps, const int* __restrict__ pd,
                    const int* __restrict__ ns, const int* __restrict__ nd,
                    const _Float16* __restrict__ Wp1t, const float* __restrict__ bp1,
                    const _Float16* __restrict__ Wp2t, const float* __restrict__ bp2,
                    const float* __restrict__ Wp3, const float* __restrict__ bp3,
                    float* __restrict__ out) {
    __shared__ _Float16 z1s[4][16 * 64];   // per-wave f16 activations
    __shared__ float    z2s[4][16 * 64];   // per-wave f32 activations
    const int tilesPerSet = P_CAND / 16;   // 6250
    int w = threadIdx.x >> 5;
    int tile = blockIdx.x * 4 + w;
    if (tile >= 2 * tilesPerSet) return;
    bool isNeg = tile >= tilesPerSet;
    int t = isNeg ? tile - tilesPerSet : tile;
    const int* S = isNeg ? ns : ps;
    const int* D = isNeg ? nd : pd;

    int lane = threadIdx.x & 31;
    int half = lane >> 4, r = lane & 15;
    int c = t * 16 + r;
    int si = S[c], di = D[c];

    // layer 1: z1 = relu((hs*hd) @ Wp1 + bp1)     [16 x 64], K=128
    v16h a1[4];
#pragma unroll
    for (int kc = 0; kc < 4; ++kc) {
        int k0 = kc * 32 + half * 8;
        a1[kc] = load_a_prod(h + si * HDIM + k0, h + di * HDIM + k0);
    }
#pragma unroll
    for (int nt = 0; nt < 4; ++nt) {
        int col = nt * 16 + r;
        v8f acc = {};
#pragma unroll
        for (int kc = 0; kc < 4; ++kc) {
            int k0 = kc * 32 + half * 8;
            v16h b = load_b_f16(Wp1t + col * 128 + k0);
            acc = __builtin_amdgcn_wmma_f32_16x16x32_f16(false, a1[kc], false, b,
                                                         (short)0, acc, false, false);
        }
        float bias = bp1[col];
#pragma unroll
        for (int v = 0; v < 8; ++v) {
            float z = acc[v] + bias;
            z = z > 0.f ? z : 0.f;
            z1s[w][(v + half * 8) * 64 + col] = (_Float16)z;   // row = candidate
        }
    }

    // layer 2: z2 = relu(z1 @ Wp2 + bp2)          [16 x 64], K=64
    // same-wave DS ops are in-order -> no barrier needed
    v16h a2[2];
#pragma unroll
    for (int kc = 0; kc < 2; ++kc) {
        int k0 = kc * 32 + half * 8;
        const _Float16* zp = &z1s[w][r * 64 + k0];
        a2[kc] = load_b_f16(zp);       // same packing: elems 0..7 = k0.., 8..15 = k0+16..
    }
#pragma unroll
    for (int nt = 0; nt < 4; ++nt) {
        int col = nt * 16 + r;
        v8f acc = {};
#pragma unroll
        for (int kc = 0; kc < 2; ++kc) {
            int k0 = kc * 32 + half * 8;
            v16h b = load_b_f16(Wp2t + col * 64 + k0);
            acc = __builtin_amdgcn_wmma_f32_16x16x32_f16(false, a2[kc], false, b,
                                                         (short)0, acc, false, false);
        }
        float bias = bp2[col];
#pragma unroll
        for (int v = 0; v < 8; ++v) {
            float z = acc[v] + bias;
            z2s[w][(v + half * 8) * 64 + col] = z > 0.f ? z : 0.f;
        }
    }

    // layer 3: out[c] = z2[c,:] . Wp3 + bp3  (split 64 cols across the 2 halves)
    float part = 0.f;
    int jb = half * 32;
#pragma unroll
    for (int j = 0; j < 32; ++j)
        part += z2s[w][r * 64 + jb + j] * Wp3[jb + j];
    part += __shfl_xor(part, 16, 32);
    if (half == 0) {
        out[(isNeg ? P_CAND : 0) + t * 16 + r] = part + bp3[0];
    }
}

// ---------------------------------------------------------------------------
extern "C" void kernel_launch(void* const* d_in, const int* in_sizes, int n_in,
                              void* d_out, int out_size, void* d_ws, size_t ws_size,
                              hipStream_t stream) {
    const float* x        = (const float*)d_in[0];
    const int*   src      = (const int*)  d_in[1];
    const int*   dst      = (const int*)  d_in[2];
    const int*   pos_src  = (const int*)  d_in[3];
    const int*   pos_dst  = (const int*)  d_in[4];
    const int*   neg_src  = (const int*)  d_in[5];
    const int*   neg_dst  = (const int*)  d_in[6];
    const float* W0       = (const float*)d_in[7];
    const float* attn0    = (const float*)d_in[8];
    const float* W1       = (const float*)d_in[9];
    const float* attn1    = (const float*)d_in[10];
    const float* Wp1      = (const float*)d_in[11];
    const float* bp1      = (const float*)d_in[12];
    const float* Wp2      = (const float*)d_in[13];
    const float* bp2      = (const float*)d_in[14];
    const float* Wp3      = (const float*)d_in[15];
    const float* bp3      = (const float*)d_in[16];

    char* ws = (char*)d_ws;
    size_t off = 0;
    auto carve = [&](size_t bytes) -> char* {
        char* p = ws + off;
        off += (bytes + 255) & ~(size_t)255;
        return p;
    };
    float*     feat  = (float*)    carve((size_t)N_NODES * HDIM * 4);
    float*     h0    = (float*)    carve((size_t)N_NODES * HDIM * 4);
    float*     h1    = (float*)    carve((size_t)N_NODES * HDIM * 4);
    float*     rst   = (float*)    carve((size_t)N_NODES * HDIM * 4);
    float*     score = (float*)    carve((size_t)E_EDGES * 2 * 4);
    int*       mOrd  = (int*)      carve((size_t)N_NODES * 2 * 4);
    float*     ssum  = (float*)    carve((size_t)N_NODES * 2 * 4);
    _Float16*  W0t   = (_Float16*) carve(128 * 128 * 2);
    _Float16*  W1t   = (_Float16*) carve(128 * 128 * 2);
    _Float16*  Wp1t  = (_Float16*) carve(64 * 128 * 2);
    _Float16*  Wp2t  = (_Float16*) carve(64 * 64 * 2);

    prep_weights<<<(128 * 128 + 255) / 256, 256, 0, stream>>>(
        W0, W1, Wp1, Wp2, W0t, W1t, Wp1t, Wp2t);

    const int gemmBlocks  = (N_NODES / 16) * 8 / 8;     // 3125 (8 waves/block)
    const int edgeBlocks  = E_EDGES / 8;                // wave per edge, 8/block
    const int ehBlocks    = (E_EDGES * 2 + 255) / 256;
    const int nodeBlocks  = (N_NODES * HDIM + 255) / 256;

    // ---------------- layer 0 (no residual) ----------------
    hipMemsetAsync(rst,  0,    (size_t)N_NODES * HDIM * 4, stream);
    hipMemsetAsync(ssum, 0,    (size_t)N_NODES * 2 * 4,    stream);
    hipMemsetAsync(mOrd, 0x80, (size_t)N_NODES * 2 * 4,    stream);  // ~ -inf ordered
    gemm128_wmma<<<gemmBlocks, 256, 0, stream>>>(x, W0t, feat, N_NODES);
    edge_score_kernel<<<edgeBlocks, 256, 0, stream>>>(feat, src, dst, attn0, score, mOrd);
    edge_expsum_kernel<<<ehBlocks, 256, 0, stream>>>(dst, mOrd, score, ssum);
    edge_aggregate_kernel<<<edgeBlocks, 256, 0, stream>>>(feat, src, dst, score, ssum, rst);
    finalize_kernel<<<nodeBlocks, 256, 0, stream>>>(rst, x, h0, 0, N_NODES * HDIM);

    // ---------------- layer 1 (residual) ----------------
    hipMemsetAsync(rst,  0,    (size_t)N_NODES * HDIM * 4, stream);
    hipMemsetAsync(ssum, 0,    (size_t)N_NODES * 2 * 4,    stream);
    hipMemsetAsync(mOrd, 0x80, (size_t)N_NODES * 2 * 4,    stream);
    gemm128_wmma<<<gemmBlocks, 256, 0, stream>>>(h0, W1t, feat, N_NODES);
    edge_score_kernel<<<edgeBlocks, 256, 0, stream>>>(feat, src, dst, attn1, score, mOrd);
    edge_expsum_kernel<<<ehBlocks, 256, 0, stream>>>(dst, mOrd, score, ssum);
    edge_aggregate_kernel<<<edgeBlocks, 256, 0, stream>>>(feat, src, dst, score, ssum, rst);
    finalize_kernel<<<nodeBlocks, 256, 0, stream>>>(rst, h0, h1, 1, N_NODES * HDIM);

    // ---------------- predictor ----------------
    const int predBlocks = (2 * (P_CAND / 16) + 3) / 4;   // 4 waves/block
    predictor_wmma<<<predBlocks, 128, 0, stream>>>(
        h1, pos_src, pos_dst, neg_src, neg_dst,
        Wp1t, bp1, Wp2t, bp2, Wp3, bp3, (float*)d_out);
}
